// CPMAntTransformerBlock_17471926960786
// MI455X (gfx1250) — matrix operand order repaired and checked by reading, not compiled, not run on hardware
//
#include <hip/hip_runtime.h>
#include <math.h>

typedef __attribute__((ext_vector_type(16))) __bf16        v16bf;
typedef __attribute__((ext_vector_type(8)))  float         v8f;
typedef __attribute__((ext_vector_type(4)))  unsigned int  v4u;
typedef __attribute__((ext_vector_type(8)))  int           v8i;
typedef __attribute__((ext_vector_type(4)))  int           v4i;
typedef unsigned short u16;

union FragB16 { uint4 u4[2]; v16bf v; };

__device__ __forceinline__ u16 f32_to_bf16(float f) {
    unsigned int u = __float_as_uint(f);
    unsigned int r = u + 0x7FFFu + ((u >> 16) & 1u);
    return (u16)(r >> 16);
}

__device__ __forceinline__ unsigned pack_bf16(float lo, float hi) {
    return (unsigned)f32_to_bf16(lo) | ((unsigned)f32_to_bf16(hi) << 16);
}

__device__ __forceinline__ v8f wmma_bf16(v16bf a, v16bf b, v8f c) {
    return __builtin_amdgcn_wmma_f32_16x16x32_bf16(false, a, false, b, (short)0, c, false, false);
}

// ---------------------------------------------------------------------------
// Tensor Data Mover: async DMA of a 2D bf16 tile (tileK x tileRows, row-major,
// row stride = ldK elements) from global into LDS. LDS rows are padded:
// 64B of data then 16B pad (pad_interval = 16 DWORDs, pad_amount = 4 DWORDs),
// matching the 40-halfword padded LDS row used by the WMMA fragment loads.
// Issue from ONE wave only; TDM ignores EXEC. Completion: s_wait_tensorcnt.
// ---------------------------------------------------------------------------
__device__ __forceinline__ void tdm_load_tile_bf16(const u16* gsrc, unsigned ldsAddr,
                                                   int ldK, int rowsTotal) {
    unsigned long long ga = (unsigned long long)(uintptr_t)gsrc;
    v4u g0 = {};
    g0[0] = 1u;                                             // count=1 (valid), user mode
    g0[1] = ldsAddr;                                        // lds_addr (bytes)
    g0[2] = (unsigned)(ga & 0xFFFFFFFFu);                   // global_addr[31:0]
    g0[3] = (unsigned)((ga >> 32) & 0x01FFFFFFu) | (2u << 30);  // addr[56:32] | type=2

    v8i g1 = {};
    g1[0] = (int)((1u << 16)        // data_size = 2 bytes
                | (1u << 20)        // pad_enable
                | (3u << 22)        // pad_interval code 3 -> 16 DWORDs (64B)
                | (3u << 25));      // pad_amount  code 3 -> 4 DWORDs (16B)
    g1[1] = (int)(((unsigned)ldK & 0xFFFFu) << 16);                     // tensor_dim0 lo16
    g1[2] = (int)((((unsigned)ldK >> 16) & 0xFFFFu)
                | (((unsigned)rowsTotal & 0xFFFFu) << 16));             // dim0 hi | dim1 lo
    g1[3] = (int)((((unsigned)rowsTotal >> 16) & 0xFFFFu) | (32u << 16)); // dim1 hi | tile_dim0=32
    g1[4] = (int)128u;                                                  // tile_dim1=128, tile_dim2=0
    g1[5] = (int)ldK;                                                   // tensor_dim0_stride lo32
    g1[6] = 0;                                                          // stride hi | dim1_stride lo
    g1[7] = 0;

    v4i z4 = {};
#if defined(__clang_major__) && (__clang_major__ >= 23)
    v8i z8 = {};
    __builtin_amdgcn_tensor_load_to_lds(g0, g1, z4, z4, z8, 0);
#else
    __builtin_amdgcn_tensor_load_to_lds(g0, g1, z4, z4, 0);
#endif
}

// ---------------------------------------------------------------------------
// RMSNorm: fp32 in -> bf16 out (one block per row)
// ---------------------------------------------------------------------------
__global__ __launch_bounds__(256)
void rmsnorm_bf16(const float* __restrict__ in, const float* __restrict__ w,
                  u16* __restrict__ out, int D) {
    __shared__ float red[256];
    int row = blockIdx.x, tid = threadIdx.x;
    const float* x = in + (size_t)row * D;
    float s = 0.f;
    for (int c = tid; c < D; c += 256) { float v = x[c]; s += v * v; }
    red[tid] = s; __syncthreads();
    for (int off = 128; off > 0; off >>= 1) {
        if (tid < off) red[tid] += red[tid + off];
        __syncthreads();
    }
    float rs = rsqrtf(red[0] / (float)D + 1e-6f);
    for (int c = tid; c < D; c += 256)
        out[(size_t)row * D + c] = f32_to_bf16(x[c] * rs * w[c]);
}

// ---------------------------------------------------------------------------
// GEMM: C[M,N] = A_bf16[M,K] @ B_f32[K,N]  (B converted to bf16 in LDS stage)
// Block tile 128x128, K-tile 32 (== WMMA K). 256 threads = 8 waves,
// wave tile 64x32 -> 4x2 WMMA accumulators. A tile staged by the TDM.
// ---------------------------------------------------------------------------
enum { EPI_QKV = 0, EPI_F32 = 1, EPI_ADD = 2, EPI_GELU_MUL = 3 };

template <int EPI>
__global__ __launch_bounds__(256)
void gemm_bf16(const u16* __restrict__ A, const float* __restrict__ B,
               int M, int N, int K,
               const float* __restrict__ aux,   // residual (ADD) or t0 (GELU_MUL)
               float* __restrict__ outF,        // F32 / ADD
               u16* __restrict__ outH,          // QKV / GELU_MUL
               int seqLen) {
    __shared__ u16 As[128 * 40];  // [row][k], pad 32->40 (row = 80B, 16B-aligned)
    __shared__ u16 Bs[128 * 40];  // [n][k] transposed, same padding

    const int tid   = threadIdx.x;
    const int lane  = tid & 31;
    const int wid   = tid >> 5;
    const int waveM = wid & 1;    // 2 waves along M
    const int waveN = wid >> 1;   // 4 waves along N
    const int mrow  = lane & 15;
    const int hi    = lane >> 4;
    const int bM = blockIdx.y * 128, bN = blockIdx.x * 128;
    const unsigned AsAddr = (unsigned)(uintptr_t)(&As[0]);

    v8f acc[4][2];
#pragma unroll
    for (int i = 0; i < 4; ++i)
#pragma unroll
        for (int j = 0; j < 2; ++j)
#pragma unroll
            for (int r = 0; r < 8; ++r) acc[i][j][r] = 0.f;

    for (int k0 = 0; k0 < K; k0 += 32) {
        // --- A tile via Tensor Data Mover (wave 0 issues; branch is uniform) ---
        if (wid == 0)
            tdm_load_tile_bf16(A + (size_t)bM * K + k0, AsAddr, K, M);

        // --- stage B tile (fp32 -> bf16, transposed to [n][k], paired-k pack) ---
        {
            int kp = (tid & 15) * 2;   // even k within tile
            int nb = tid >> 4;         // 0..15 -> 8 n's each
            const float* s0 = B + (size_t)(k0 + kp) * N + bN + nb * 8;
            const float* s1 = s0 + N;
#pragma unroll
            for (int j = 0; j < 8; j += 4) {
                float4 f0 = *(const float4*)(s0 + j);
                float4 f1 = *(const float4*)(s1 + j);
                *(unsigned*)&Bs[(nb * 8 + j + 0) * 40 + kp] = pack_bf16(f0.x, f1.x);
                *(unsigned*)&Bs[(nb * 8 + j + 1) * 40 + kp] = pack_bf16(f0.y, f1.y);
                *(unsigned*)&Bs[(nb * 8 + j + 2) * 40 + kp] = pack_bf16(f0.z, f1.z);
                *(unsigned*)&Bs[(nb * 8 + j + 3) * 40 + kp] = pack_bf16(f0.w, f1.w);
            }
            if (k0 + 32 < K)  // gfx1250 global_prefetch_b8 for next K tile
                __builtin_prefetch((const void*)(s0 + 32 * N), 0, 1);
        }
        if (wid == 0) __builtin_amdgcn_s_wait_tensorcnt(0);
        __syncthreads();

        // --- fragments + 8 WMMAs ---
        v16bf af[4], bf[2];
#pragma unroll
        for (int i = 0; i < 4; ++i) {
            int r = waveM * 64 + i * 16 + mrow;
            FragB16 f;
            f.u4[0] = *(const uint4*)(&As[r * 40 + hi * 8]);        // K = hi*8 .. +7
            f.u4[1] = *(const uint4*)(&As[r * 40 + 16 + hi * 8]);   // K = 16+hi*8 .. +7
            af[i] = f.v;
        }
#pragma unroll
        for (int j = 0; j < 2; ++j) {
            int n = waveN * 32 + j * 16 + mrow;
            FragB16 f;
            f.u4[0] = *(const uint4*)(&Bs[n * 40 + hi * 8]);        // K = hi*8 .. +7
            f.u4[1] = *(const uint4*)(&Bs[n * 40 + 16 + hi * 8]);   // K = 16+hi*8 .. +7
            bf[j] = f.v;
        }
#pragma unroll
        for (int i = 0; i < 4; ++i)
#pragma unroll
            for (int j = 0; j < 2; ++j)
                acc[i][j] = wmma_bf16(af[i], bf[j], acc[i][j]);
        __syncthreads();
    }

    // --- epilogue (C layout: elem r of lane -> row r + 8*hi, col = lane&15) ---
#pragma unroll
    for (int i = 0; i < 4; ++i)
#pragma unroll
        for (int j = 0; j < 2; ++j) {
            int nG = bN + waveN * 32 + j * 16 + mrow;
#pragma unroll
            for (int r = 0; r < 8; ++r) {
                int mG = bM + waveM * 64 + i * 16 + r + 8 * hi;
                float v = acc[i][j][r];
                size_t idx = (size_t)mG * N + nG;
                if (EPI == EPI_F32) {
                    outF[idx] = v;
                } else if (EPI == EPI_ADD) {
                    outF[idx] = v + aux[idx];
                } else if (EPI == EPI_QKV) {
                    int head = nG >> 7, dh = nG & 127;
                    outH[((size_t)head * seqLen + mG) * 128 + dh] = f32_to_bf16(v);
                } else {  // EPI_GELU_MUL: out = bf16(gelu(t0) * acc)
                    float g  = aux[idx];
                    float ge = 0.5f * g * (1.0f + erff(g * 0.70710678118654752f));
                    outH[idx] = f32_to_bf16(ge * v);
                }
            }
        }
}

// ---------------------------------------------------------------------------
// Flash attention: grid (S/64 q-tiles, H heads), 128 threads = 4 waves,
// each wave owns 16 query rows. Q/K/V in bf16 head-major [H][S][128].
// ---------------------------------------------------------------------------
__global__ __launch_bounds__(128)
void attn_kernel(const u16* __restrict__ qh, const u16* __restrict__ khm,
                 const u16* __restrict__ vh, const float* __restrict__ bias,
                 const unsigned char* __restrict__ mask,
                 u16* __restrict__ out, int S) {
    __shared__ u16 Ks[64 * 128];      // [key][dh]
    __shared__ u16 Vt[128 * 72];      // [dh][key], pad 64->72 (row = 144B)
    __shared__ u16 Pw[4][16 * 72];    // per-wave P tile [qrow][key], pad 64->72

    const int h = blockIdx.y, qt = blockIdx.x;
    const int tid = threadIdx.x, wid = tid >> 5, lane = tid & 31;
    const int mrow = lane & 15, hi = lane >> 4;
    const int qbase = qt * 64 + wid * 16;
    const float scale = 0.088388347648318447f;  // 1/sqrt(128)

    // Q fragments (A layout): 4 k-steps over Dh=128
    v16bf qf[4];
    const u16* qrow = qh + ((size_t)h * S + qbase + mrow) * 128;
#pragma unroll
    for (int t = 0; t < 4; ++t) {
        FragB16 f;
        f.u4[0] = *(const uint4*)(qrow + t * 32 + hi * 8);
        f.u4[1] = *(const uint4*)(qrow + t * 32 + 16 + hi * 8);
        qf[t] = f.v;
    }

    float mi[8], li[8]; v8f o[8];
#pragma unroll
    for (int r = 0; r < 8; ++r) { mi[r] = -__builtin_inff(); li[r] = 0.f; }
#pragma unroll
    for (int c = 0; c < 8; ++c)
#pragma unroll
        for (int r = 0; r < 8; ++r) o[c][r] = 0.f;

    for (int kt = 0; kt < S; kt += 64) {
        // stage K tile row-major (B-frag reads are contiguous in dh)
        {
            int row = tid >> 1, hf = tid & 1;
            const uint4* src = (const uint4*)(khm + ((size_t)h * S + kt + row) * 128 + hf * 64);
            uint4* dst = (uint4*)(&Ks[row * 128 + hf * 64]);
#pragma unroll
            for (int u = 0; u < 8; ++u) dst[u] = src[u];
        }
        // stage V tile transposed [dh][key]: thread handles a key PAIR so each
        // LDS write is one packed b32 (keys key2,key2+1 are adjacent halfwords)
        {
            int key2 = (tid & 31) * 2;
            int dh0  = (tid >> 5) * 32;
            const u16* s0 = vh + ((size_t)h * S + kt + key2) * 128 + dh0;
            const u16* s1 = s0 + 128;
#pragma unroll
            for (int g = 0; g < 4; ++g) {
                uint4 a = *(const uint4*)(s0 + g * 8);
                uint4 b = *(const uint4*)(s1 + g * 8);
                unsigned aw[4] = {a.x, a.y, a.z, a.w};
                unsigned bw[4] = {b.x, b.y, b.z, b.w};
#pragma unroll
                for (int q = 0; q < 4; ++q) {
                    unsigned lo = (aw[q] & 0xFFFFu) | (bw[q] << 16);
                    unsigned hh = (aw[q] >> 16) | (bw[q] & 0xFFFF0000u);
                    *(unsigned*)&Vt[(dh0 + g * 8 + q * 2 + 0) * 72 + key2] = lo;
                    *(unsigned*)&Vt[(dh0 + g * 8 + q * 2 + 1) * 72 + key2] = hh;
                }
            }
        }
        __syncthreads();

        // S = Q K^T : 4 n-frags (64 keys) x 4 k-steps
        v8f sacc[4];
#pragma unroll
        for (int j = 0; j < 4; ++j)
#pragma unroll
            for (int r = 0; r < 8; ++r) sacc[j][r] = 0.f;
#pragma unroll
        for (int t = 0; t < 4; ++t)
#pragma unroll
            for (int j = 0; j < 4; ++j) {
                int n = j * 16 + mrow;
                FragB16 f;
                f.u4[0] = *(const uint4*)(&Ks[n * 128 + t * 32 + hi * 8]);
                f.u4[1] = *(const uint4*)(&Ks[n * 128 + t * 32 + 16 + hi * 8]);
                sacc[j] = wmma_bf16(qf[t], f.v, sacc[j]);
            }

        // scale + bias + mask
        float sv[4][8];
#pragma unroll
        for (int j = 0; j < 4; ++j) {
            int key = kt + j * 16 + mrow;
#pragma unroll
            for (int r = 0; r < 8; ++r) {
                int q = qbase + r + 8 * hi;
                float s = sacc[j][r] * scale + bias[((size_t)h * S + q) * S + key];
                if (!mask[(size_t)q * S + key]) s = -__builtin_inff();
                sv[j][r] = s;
            }
        }

        // online softmax (row = r + 8*hi, reduce across the 16-lane group)
#pragma unroll
        for (int r = 0; r < 8; ++r) {
            float mx = sv[0][r];
#pragma unroll
            for (int j = 1; j < 4; ++j) mx = fmaxf(mx, sv[j][r]);
#pragma unroll
            for (int d = 1; d < 16; d <<= 1) mx = fmaxf(mx, __shfl_xor(mx, d, 32));
            float nm = fmaxf(mi[r], mx);
            float corr = __expf(mi[r] - nm);
            float rs = 0.f;
#pragma unroll
            for (int j = 0; j < 4; ++j) {
                float p = __expf(sv[j][r] - nm);
                sv[j][r] = p; rs += p;
            }
#pragma unroll
            for (int d = 1; d < 16; d <<= 1) rs += __shfl_xor(rs, d, 32);
            li[r] = li[r] * corr + rs;
            mi[r] = nm;
#pragma unroll
            for (int c = 0; c < 8; ++c) o[c][r] *= corr;
        }

        // P -> LDS (re-layout C-frag -> A-frag via wave-private tile)
#pragma unroll
        for (int j = 0; j < 4; ++j)
#pragma unroll
            for (int r = 0; r < 8; ++r)
                Pw[wid][(r + 8 * hi) * 72 + j * 16 + mrow] = f32_to_bf16(sv[j][r]);
        __syncthreads();

        // O += P V : 2 k-steps over 64 keys x 8 dh-chunks
#pragma unroll
        for (int t = 0; t < 2; ++t) {
            FragB16 fa;
            fa.u4[0] = *(const uint4*)(&Pw[wid][mrow * 72 + t * 32 + hi * 8]);
            fa.u4[1] = *(const uint4*)(&Pw[wid][mrow * 72 + t * 32 + 16 + hi * 8]);
#pragma unroll
            for (int c = 0; c < 8; ++c) {
                int n = c * 16 + mrow;
                FragB16 fb;
                fb.u4[0] = *(const uint4*)(&Vt[n * 72 + t * 32 + hi * 8]);
                fb.u4[1] = *(const uint4*)(&Vt[n * 72 + t * 32 + 16 + hi * 8]);
                o[c] = wmma_bf16(fa.v, fb.v, o[c]);
            }
        }
        __syncthreads();
    }

    // write attn output bf16 row-major [S][H*128]
#pragma unroll
    for (int c = 0; c < 8; ++c)
#pragma unroll
        for (int r = 0; r < 8; ++r) {
            int q = qbase + r + 8 * hi;
            out[(size_t)q * 4096 + h * 128 + c * 16 + mrow] = f32_to_bf16(o[c][r] / li[r]);
        }
}

// ---------------------------------------------------------------------------
// Launch
// ---------------------------------------------------------------------------
extern "C" void kernel_launch(void* const* d_in, const int* in_sizes, int n_in,
                              void* d_out, int out_size, void* d_ws, size_t ws_size,
                              hipStream_t stream) {
    (void)in_sizes; (void)n_in; (void)out_size; (void)ws_size;
    const float*         hidden = (const float*)d_in[0];
    const unsigned char* mask   = (const unsigned char*)d_in[1];
    const float*         pbias  = (const float*)d_in[2];
    const float*         ln1    = (const float*)d_in[3];
    const float*         wq     = (const float*)d_in[4];
    const float*         wk     = (const float*)d_in[5];
    const float*         wv     = (const float*)d_in[6];
    const float*         wo     = (const float*)d_in[7];
    const float*         ln2    = (const float*)d_in[8];
    const float*         w0     = (const float*)d_in[9];
    const float*         w1     = (const float*)d_in[10];
    const float*         wout   = (const float*)d_in[11];
    float* out = (float*)d_out;

    const int S = 2048, D = 4096, FF = 10240, H = 32;

    char* ws = (char*)d_ws;
    size_t off = 0;
    auto alloc = [&](size_t bytes) { char* p = ws + off; off += (bytes + 255) & ~(size_t)255; return p; };
    u16*   xb  = (u16*)  alloc((size_t)S * D * 2);   // normed x, bf16
    u16*   qb  = (u16*)  alloc((size_t)S * D * 2);   // Q head-major bf16
    u16*   kb  = (u16*)  alloc((size_t)S * D * 2);   // K head-major bf16
    u16*   vb  = (u16*)  alloc((size_t)S * D * 2);   // V head-major bf16
    u16*   ab  = (u16*)  alloc((size_t)S * D * 2);   // attn out bf16
    float* h2  = (float*)alloc((size_t)S * D * 4);   // hidden after attn residual
    u16*   yb  = (u16*)  alloc((size_t)S * D * 2);   // normed y bf16
    float* t0  = (float*)alloc((size_t)S * FF * 4);  // y @ w0 fp32
    u16*   ffb = (u16*)  alloc((size_t)S * FF * 2);  // gated ff bf16

    rmsnorm_bf16<<<S, 256, 0, stream>>>(hidden, ln1, xb, D);

    dim3 gQKV(D / 128, S / 128);
    gemm_bf16<EPI_QKV><<<gQKV, 256, 0, stream>>>(xb, wq, S, D, D, nullptr, nullptr, qb, S);
    gemm_bf16<EPI_QKV><<<gQKV, 256, 0, stream>>>(xb, wk, S, D, D, nullptr, nullptr, kb, S);
    gemm_bf16<EPI_QKV><<<gQKV, 256, 0, stream>>>(xb, wv, S, D, D, nullptr, nullptr, vb, S);

    attn_kernel<<<dim3(S / 64, H), 128, 0, stream>>>(qb, kb, vb, pbias, mask, ab, S);

    gemm_bf16<EPI_ADD><<<dim3(D / 128, S / 128), 256, 0, stream>>>(ab, wo, S, D, D, hidden, h2, nullptr, S);

    rmsnorm_bf16<<<S, 256, 0, stream>>>(h2, ln2, yb, D);

    dim3 gFF(FF / 128, S / 128);
    gemm_bf16<EPI_F32><<<gFF, 256, 0, stream>>>(yb, w0, S, FF, D, nullptr, t0, nullptr, S);
    gemm_bf16<EPI_GELU_MUL><<<gFF, 256, 0, stream>>>(yb, w1, S, FF, D, t0, nullptr, ffb, S);

    gemm_bf16<EPI_ADD><<<dim3(D / 128, S / 128), 256, 0, stream>>>(ffb, wout, S, D, FF, h2, out, nullptr, S);
}
